// GPT_10299331576326
// MI455X (gfx1250) — matrix operand order repaired and checked by reading, not compiled, not run on hardware
//
#include <hip/hip_runtime.h>

// ---------------------------------------------------------------------------
// GPT-2 small forward (L=12,H=12,C=768,V=50257,B=2,T=1024) for gfx1250.
// All GEMMs via v_wmma_f32_16x16x32_bf16 (bf16 inputs, fp32 accum).
// GEMM: 256x64 block tile, 8 waves x (32x64) wave tiles, TRIPLE-buffered LDS
// B tile staged with global_load_async_to_lds_b128 prefetch pipeline:
//   [wait asynccnt 0] -> barrier -> issue async prefetch(next) -> 8x WMMA(cur)
// so the global->LDS latency of tile i+1 hides behind the WMMAs of tile i.
// ---------------------------------------------------------------------------

typedef __bf16 bf16;
typedef __attribute__((ext_vector_type(16))) __bf16 bf16x16;
typedef __attribute__((ext_vector_type(8)))  __bf16 bf16x8;
typedef __attribute__((ext_vector_type(8)))  float  f32x8;
typedef __attribute__((__vector_size__(16))) int    i32x4v;

static constexpr int Lc = 12, Hc = 12, Cc = 768, Vc = 50257, Bc = 2, Tc = 1024, Dc = 64;

#ifndef __has_builtin
#define __has_builtin(x) 0
#endif

#if defined(__HIP_DEVICE_COMPILE__) && \
    __has_builtin(__builtin_amdgcn_global_load_async_to_lds_b128) && \
    __has_builtin(__builtin_amdgcn_s_wait_asynccnt)
#define USE_ASYNC_LDS 1
#else
#define USE_ASYNC_LDS 0
#endif

#define GLOBAL_AS __attribute__((address_space(1)))
#define LDS_AS    __attribute__((address_space(3)))

// ---------------------------------------------------------------------------
// FIRE bias: bias[i,j] = MLP(log(c*(i-j)+1)/log(c*i+1)) on tril, -1e30 above.
// ---------------------------------------------------------------------------
__global__ __launch_bounds__(256) void fire_bias_kernel(
    const float* __restrict__ log_c,
    const float* __restrict__ v1w, const float* __restrict__ v1b,
    const float* __restrict__ v2w, const float* __restrict__ v2b,
    const float* __restrict__ v3w, const float* __restrict__ v3b,
    float* __restrict__ out) {
  int gid = blockIdx.x * 256 + threadIdx.x;
  if (gid >= Tc * Tc) return;
  int i = gid / Tc, j = gid % Tc;
  float c = expf(log_c[0]);
  float fi = (float)(i + 1), fj = (float)(j + 1);
  bool tril = fi >= fj;
  float arg = tril ? (c * (fi - fj) + 1.0f) : 1.0f;
  float pv = logf(arg) / logf(c * fi + 1.0f);
  float h1[32], h2[32];
#pragma unroll
  for (int o = 0; o < 32; ++o) {
    float s = pv * v1w[o] + v1b[o];
    h1[o] = s > 0.0f ? s : 0.0f;
  }
#pragma unroll 4
  for (int o = 0; o < 32; ++o) {
    float s = v2b[o];
    for (int k = 0; k < 32; ++k) s += h1[k] * v2w[k * 32 + o];
    h2[o] = s > 0.0f ? s : 0.0f;
  }
  float f = v3b[0];
#pragma unroll
  for (int o = 0; o < 32; ++o) f += h2[o] * v3w[o];
  out[gid] = tril ? f : -1e30f;
}

// ---------------------------------------------------------------------------
// fp32 -> bf16 flat convert (wte: lm-head B matrix is wte itself, N x K)
// ---------------------------------------------------------------------------
__global__ __launch_bounds__(256) void cvt_bf16_kernel(const float* __restrict__ in,
                                                       bf16* __restrict__ out, long long n) {
  long long i = (long long)blockIdx.x * 256 + threadIdx.x;
  if (i < n) out[i] = (bf16)in[i];
}

// ---------------------------------------------------------------------------
// Transpose + convert: W (K x N fp32) -> Wt (N x K bf16). K,N multiples of 32.
// ---------------------------------------------------------------------------
__global__ __launch_bounds__(256) void transpose_cvt_kernel(const float* __restrict__ W,
                                                            bf16* __restrict__ Wt,
                                                            int K, int N) {
  __shared__ float tile[32][33];
  int nn = blockIdx.x * 32, kk = blockIdx.y * 32;
  int tx = threadIdx.x & 31, ty = threadIdx.x >> 5;
#pragma unroll
  for (int i = 0; i < 4; ++i)
    tile[ty + i * 8][tx] = W[(long long)(kk + ty + i * 8) * N + nn + tx];
  __syncthreads();
#pragma unroll
  for (int i = 0; i < 4; ++i)
    Wt[(long long)(nn + ty + i * 8) * K + kk + tx] = (bf16)tile[tx][ty + i * 8];
}

// ---------------------------------------------------------------------------
// Embedding gather
// ---------------------------------------------------------------------------
__global__ __launch_bounds__(256) void embed_kernel(const int* __restrict__ idx,
                                                    const float* __restrict__ wte,
                                                    float* __restrict__ x) {
  int gid = blockIdx.x * 256 + threadIdx.x;  // over B*T*C
  int c = gid % Cc, row = gid / Cc;
  x[gid] = wte[(long long)idx[row] * Cc + c];
}

// ---------------------------------------------------------------------------
// LayerNorm (fp32 in) -> bf16 out. One block per row, C=768.
// ---------------------------------------------------------------------------
__global__ __launch_bounds__(256) void ln_kernel(const float* __restrict__ x,
                                                 const float* __restrict__ g,
                                                 const float* __restrict__ b,
                                                 bf16* __restrict__ out) {
  __shared__ float red[256];
  int row = blockIdx.x, tid = threadIdx.x;
  const float* xr = x + (long long)row * Cc;
  float s = 0.0f;
  for (int c = tid; c < Cc; c += 256) s += xr[c];
  red[tid] = s; __syncthreads();
  for (int st = 128; st > 0; st >>= 1) { if (tid < st) red[tid] += red[tid + st]; __syncthreads(); }
  float mean = red[0] / (float)Cc; __syncthreads();
  float v = 0.0f;
  for (int c = tid; c < Cc; c += 256) { float d = xr[c] - mean; v += d * d; }
  red[tid] = v; __syncthreads();
  for (int st = 128; st > 0; st >>= 1) { if (tid < st) red[tid] += red[tid + st]; __syncthreads(); }
  float rinv = rsqrtf(red[0] / (float)Cc + 1e-5f);
  for (int c = tid; c < Cc; c += 256)
    out[(long long)row * Cc + c] = (bf16)((xr[c] - mean) * rinv * g[c] + b[c]);
}

// ---------------------------------------------------------------------------
// qkv (B*T x 3C fp32) -> q,k [B,H,T,D] bf16, vT [B,H,D,T] bf16
// ---------------------------------------------------------------------------
__global__ __launch_bounds__(256) void repack_qkv_kernel(const float* __restrict__ qkv,
                                                         bf16* __restrict__ q,
                                                         bf16* __restrict__ k,
                                                         bf16* __restrict__ vt) {
  int gid = blockIdx.x * 256 + threadIdx.x;  // over B*T*C
  int c = gid % Cc, row = gid / Cc;
  int b = row / Tc, t = row % Tc;
  int h = c / Dc, d = c % Dc;
  const float* base = qkv + (long long)row * 3 * Cc;
  long long qi = (((long long)(b * Hc + h)) * Tc + t) * Dc + d;
  q[qi] = (bf16)base[c];
  k[qi] = (bf16)base[Cc + c];
  vt[(((long long)(b * Hc + h)) * Dc + d) * Tc + t] = (bf16)base[2 * Cc + c];
}

// ---------------------------------------------------------------------------
// Row softmax over fp32 scores -> bf16 probabilities (row length 1024).
// ---------------------------------------------------------------------------
__global__ __launch_bounds__(256) void softmax_kernel(const float* __restrict__ S,
                                                      bf16* __restrict__ P) {
  __shared__ float red[256];
  long long row = blockIdx.x;
  const float* s = S + row * (long long)Tc;
  bf16* p = P + row * (long long)Tc;
  int tid = threadIdx.x;
  float m = -3.4e38f;
  for (int i = tid; i < Tc; i += 256) m = fmaxf(m, s[i]);
  red[tid] = m; __syncthreads();
  for (int st = 128; st > 0; st >>= 1) { if (tid < st) red[tid] = fmaxf(red[tid], red[tid + st]); __syncthreads(); }
  m = red[0]; __syncthreads();
  float sum = 0.0f;
  for (int i = tid; i < Tc; i += 256) sum += expf(s[i] - m);
  red[tid] = sum; __syncthreads();
  for (int st = 128; st > 0; st >>= 1) { if (tid < st) red[tid] += red[tid + st]; __syncthreads(); }
  float inv = 1.0f / red[0];
  for (int i = tid; i < Tc; i += 256) p[i] = (bf16)(expf(s[i] - m) * inv);
}

// ---------------------------------------------------------------------------
// y [B,H,T,D] fp32 -> y_bf [B,T,C] bf16 (head merge)
// ---------------------------------------------------------------------------
__global__ __launch_bounds__(256) void merge_heads_kernel(const float* __restrict__ y,
                                                          bf16* __restrict__ out) {
  int gid = blockIdx.x * 256 + threadIdx.x;  // over B*T*C
  int c = gid % Cc, row = gid / Cc;
  int b = row / Tc, t = row % Tc;
  int h = c / Dc, d = c % Dc;
  out[gid] = (bf16)y[(((long long)(b * Hc + h)) * Tc + t) * Dc + d];
}

// ---------------------------------------------------------------------------
// B-tile staging: one 16B segment per thread, async DMA to LDS if available.
// ---------------------------------------------------------------------------
template <bool BOUND>
__device__ __forceinline__ void stage_seg(const bf16* src, bool svalid, bf16* dst) {
  if (!BOUND || svalid) {
#if USE_ASYNC_LDS
    __builtin_amdgcn_global_load_async_to_lds_b128(
        (GLOBAL_AS i32x4v*)src, (LDS_AS i32x4v*)dst, 0, 0);
#else
    *(bf16x8*)dst = *(const bf16x8*)src;
#endif
  } else {
    bf16x8 z;
#pragma unroll
    for (int i = 0; i < 8; ++i) z[i] = (bf16)0.0f;
    *(bf16x8*)dst = z;
  }
}

// ---------------------------------------------------------------------------
// WMMA GEMM: Out(M x N) = A(M x K bf16, lda) @ Bt(N x K bf16, ldb)^T + epilogue
//   Block tile 256(M) x 64(N), 256 threads = 8 waves, wave tile 32x64.
//   Triple-buffered 64x32 B tile (stride 40 -> conflict-free b128 LDS reads),
//   async prefetch pipeline, one barrier per K-step.
// Epilogues: 0: +bias  1: residual+bias  2: gelu->bf16  3: *scale+bias2d
// ---------------------------------------------------------------------------
__device__ __forceinline__ float gelu_tanh(float x) {
  float u = 0.7978845608028654f * (x + 0.044715f * x * x * x);
  return 0.5f * x * (1.0f + tanhf(u));
}

template <int EPI, bool BOUND>
__global__ __launch_bounds__(256) void gemm_bf16_kernel(
    const bf16* __restrict__ A, long long sA, int lda,
    const bf16* __restrict__ Bt, long long sB, int ldb,
    float* __restrict__ Out, long long sO, int ldo,
    bf16* __restrict__ OutBf,
    const float* __restrict__ bias,
    const float* __restrict__ bias2d, int ldb2,
    const float* __restrict__ Res,
    int M, int N, int K, float scale) {
  __shared__ alignas(16) bf16 Bs[3][64 * 40];  // triple-buffered 64 x 32, stride 40

  const int bz = blockIdx.z;
  const int n0 = blockIdx.x * 64;
  const int m0 = blockIdx.y * 256;
  const int wave = threadIdx.x >> 5;
  const int lane = threadIdx.x & 31;
  const int col = lane & 15;

  const bf16* Ab = A + sA * bz;
  const bf16* Bb = Bt + sB * bz;

  f32x8 acc[2][4];
#pragma unroll
  for (int mi = 0; mi < 2; ++mi)
#pragma unroll
    for (int c = 0; c < 4; ++c)
#pragma unroll
      for (int r = 0; r < 8; ++r) acc[mi][c][r] = 0.0f;

  const int arow  = m0 + wave * 32 + col;      // first of two 16-row fragments
  const int koffA = (lane < 16) ? 0 : 8;       // A frag: low lanes K 0..7/16..23
  const int kb    = (lane < 16) ? 0 : 8;       // B frag: low lanes K 0..7/16..23

  // staging assignment: thread -> (row, 8-elem segment) of the 64x32 B tile
  const int srow = threadIdx.x >> 2;
  const int sseg = (threadIdx.x & 3) * 8;
  const int soff = srow * 40 + sseg;
  const bool svalid = !BOUND || (n0 + srow < N);
  const bf16* sptr = Bb + (long long)(n0 + srow) * ldb + sseg;

  const int steps = K >> 5;

  // prologue: stage tile 0 into buffer 0
  stage_seg<BOUND>(sptr, svalid, &Bs[0][soff]);

  int cur = 0;
  for (int it = 0; it < steps; ++it) {
    const int k0 = it << 5;
    int nxt = cur + 1; if (nxt == 3) nxt = 0;

#if USE_ASYNC_LDS
    __builtin_amdgcn_s_wait_asynccnt(0);   // my tile-it segment landed in LDS
#endif
    __syncthreads();                        // all segments landed; compute(it-1) done everywhere

    // prefetch next tile into buffer (it+1)%3; latency hides behind WMMAs below
    if (it + 1 < steps)
      stage_seg<BOUND>(sptr + (k0 + 32), svalid, &Bs[nxt][soff]);

    // --- two A fragments (rows arow, arow+16), straight from global ---
    bf16x16 af[2];
#pragma unroll
    for (int mi = 0; mi < 2; ++mi) {
      const bf16* ap = Ab + (long long)(arow + mi * 16) * lda + k0 + koffA;
      bf16x8 alo = *(const bf16x8*)ap;
      bf16x8 ahi = *(const bf16x8*)(ap + 16);
#pragma unroll
      for (int i = 0; i < 8; ++i) { af[mi][i] = alo[i]; af[mi][8 + i] = ahi[i]; }
    }

    // --- 4 col tiles x 2 row tiles = 8 WMMAs; each B fragment reused twice ---
#pragma unroll
    for (int c = 0; c < 4; ++c) {
      const bf16* bp = &Bs[cur][(c * 16 + col) * 40 + kb];
      bf16x8 blo = *(const bf16x8*)bp;
      bf16x8 bhi = *(const bf16x8*)(bp + 16);
      bf16x16 bfrag;
#pragma unroll
      for (int i = 0; i < 8; ++i) { bfrag[i] = blo[i]; bfrag[8 + i] = bhi[i]; }
#pragma unroll
      for (int mi = 0; mi < 2; ++mi)
        acc[mi][c] = __builtin_amdgcn_wmma_f32_16x16x32_bf16(
            false, af[mi], false, bfrag, (short)0, acc[mi][c], false, false);
    }
    cur = nxt;
  }

  // --- epilogue: D layout vgpr r -> row (r | r+8), lane&15 -> col ---
  const int rbase = (lane < 16) ? 0 : 8;
#pragma unroll
  for (int mi = 0; mi < 2; ++mi) {
#pragma unroll
    for (int c = 0; c < 4; ++c) {
      int n = n0 + c * 16 + col;
      if (BOUND && n >= N) continue;
      float bn = (EPI == 3) ? 0.0f : (bias ? bias[n] : 0.0f);
#pragma unroll
      for (int r = 0; r < 8; ++r) {
        int m = m0 + wave * 32 + mi * 16 + rbase + r;
        float v = acc[mi][c][r] + bn;
        long long o = (long long)m * ldo + n;
        if constexpr (EPI == 0) {
          Out[sO * bz + o] = v;
        } else if constexpr (EPI == 1) {
          Out[sO * bz + o] = Res[o] + v;
        } else if constexpr (EPI == 2) {
          OutBf[o] = (bf16)gelu_tanh(v);
        } else {  // EPI == 3
          Out[sO * bz + o] = acc[mi][c][r] * scale + bias2d[(long long)m * ldb2 + n];
        }
      }
    }
  }
}

// ---------------------------------------------------------------------------
// Host-side orchestration
// ---------------------------------------------------------------------------
extern "C" void kernel_launch(void* const* d_in, const int* in_sizes, int n_in,
                              void* d_out, int out_size, void* d_ws, size_t ws_size,
                              hipStream_t stream) {
  (void)in_sizes; (void)n_in; (void)out_size; (void)ws_size;
  const int*   idx    = (const int*)d_in[0];
  const float* wte    = (const float*)d_in[2];
  const float* log_c  = (const float*)d_in[3];
  const float* v1w    = (const float*)d_in[4];
  const float* v1b    = (const float*)d_in[5];
  const float* v2w    = (const float*)d_in[6];
  const float* v2b    = (const float*)d_in[7];
  const float* v3w    = (const float*)d_in[8];
  const float* v3b    = (const float*)d_in[9];
  const float* ln1_g  = (const float*)d_in[10];
  const float* ln1_b  = (const float*)d_in[11];
  const float* attn_w = (const float*)d_in[12];
  const float* attn_b = (const float*)d_in[13];
  const float* proj_w = (const float*)d_in[14];
  const float* proj_b = (const float*)d_in[15];
  const float* ln2_g  = (const float*)d_in[16];
  const float* ln2_b  = (const float*)d_in[17];
  const float* fc_w   = (const float*)d_in[18];
  const float* fc_b   = (const float*)d_in[19];
  const float* fc2_w  = (const float*)d_in[20];
  const float* fc2_b  = (const float*)d_in[21];
  const float* lnf_g  = (const float*)d_in[22];
  const float* lnf_b  = (const float*)d_in[23];
  float* out = (float*)d_out;

  // ---- workspace bump allocator (256B aligned) ----
  char* w = (char*)d_ws;
  size_t off = 0;
  auto take = [&](size_t bytes) -> void* {
    void* p = w + off;
    off = (off + bytes + 255) & ~(size_t)255;
    return p;
  };
  float* fire   = (float*)take((size_t)Tc * Tc * 4);
  bf16* wte_bf  = (bf16*)take((size_t)Vc * Cc * 2);
  bf16* attnT   = (bf16*)take((size_t)Lc * 3 * Cc * Cc * 2);  // per layer: (3C x C)
  bf16* projT   = (bf16*)take((size_t)Lc * Cc * Cc * 2);      // (C x C)
  bf16* fcT     = (bf16*)take((size_t)Lc * 4 * Cc * Cc * 2);  // (4C x C)
  bf16* fc2T    = (bf16*)take((size_t)Lc * Cc * 4 * Cc * 2);  // (C x 4C)
  float* x      = (float*)take((size_t)Bc * Tc * Cc * 4);
  bf16* h_bf    = (bf16*)take((size_t)Bc * Tc * Cc * 2);
  float* qkv    = (float*)take((size_t)Bc * Tc * 3 * Cc * 4);
  bf16* q_bf    = (bf16*)take((size_t)Bc * Hc * Tc * Dc * 2);
  bf16* k_bf    = (bf16*)take((size_t)Bc * Hc * Tc * Dc * 2);
  bf16* vT_bf   = (bf16*)take((size_t)Bc * Hc * Tc * Dc * 2);
  float* S      = (float*)take((size_t)Bc * Hc * Tc * Tc * 4);
  bf16* P       = (bf16*)take((size_t)Bc * Hc * Tc * Tc * 2);
  float* y      = (float*)take((size_t)Bc * Hc * Tc * Dc * 4);
  bf16* y_bf    = (bf16*)take((size_t)Bc * Tc * Cc * 2);
  bf16* u_bf    = (bf16*)take((size_t)Bc * Tc * 4 * Cc * 2);

  const int rows = Bc * Tc;            // 2048
  const int btc_blocks = rows * Cc / 256;

  // ---- prep ----
  fire_bias_kernel<<<(Tc * Tc + 255) / 256, 256, 0, stream>>>(
      log_c, v1w, v1b, v2w, v2b, v3w, v3b, fire);

  long long nwte = (long long)Vc * Cc;
  cvt_bf16_kernel<<<(unsigned)((nwte + 255) / 256), 256, 0, stream>>>(wte, wte_bf, nwte);

  for (int l = 0; l < Lc; ++l) {
    transpose_cvt_kernel<<<dim3(3 * Cc / 32, Cc / 32), 256, 0, stream>>>(
        attn_w + (size_t)l * Cc * 3 * Cc, attnT + (size_t)l * 3 * Cc * Cc, Cc, 3 * Cc);
    transpose_cvt_kernel<<<dim3(Cc / 32, Cc / 32), 256, 0, stream>>>(
        proj_w + (size_t)l * Cc * Cc, projT + (size_t)l * Cc * Cc, Cc, Cc);
    transpose_cvt_kernel<<<dim3(4 * Cc / 32, Cc / 32), 256, 0, stream>>>(
        fc_w + (size_t)l * Cc * 4 * Cc, fcT + (size_t)l * 4 * Cc * Cc, Cc, 4 * Cc);
    transpose_cvt_kernel<<<dim3(Cc / 32, 4 * Cc / 32), 256, 0, stream>>>(
        fc2_w + (size_t)l * 4 * Cc * Cc, fc2T + (size_t)l * Cc * 4 * Cc, 4 * Cc, Cc);
  }

  embed_kernel<<<btc_blocks, 256, 0, stream>>>(idx, wte, x);

  const float scale = 0.125f;  // 1/sqrt(64)

  for (int l = 0; l < Lc; ++l) {
    const bf16* awT = attnT + (size_t)l * 3 * Cc * Cc;
    const bf16* pwT = projT + (size_t)l * Cc * Cc;
    const bf16* fwT = fcT + (size_t)l * 4 * Cc * Cc;
    const bf16* f2T = fc2T + (size_t)l * Cc * 4 * Cc;

    // h = LN1(x) -> bf16
    ln_kernel<<<rows, 256, 0, stream>>>(x, ln1_g + (size_t)l * Cc, ln1_b + (size_t)l * Cc, h_bf);

    // qkv = h @ attn_w + attn_b   (M=2048, N=2304, K=768)
    gemm_bf16_kernel<0, false><<<dim3(3 * Cc / 64, rows / 256, 1), 256, 0, stream>>>(
        h_bf, 0, Cc, awT, 0, Cc, qkv, 0, 3 * Cc, nullptr,
        attn_b + (size_t)l * 3 * Cc, nullptr, 0, nullptr, rows, 3 * Cc, Cc, 1.0f);

    repack_qkv_kernel<<<btc_blocks, 256, 0, stream>>>(qkv, q_bf, k_bf, vT_bf);

    // S = Q @ K^T * scale + fire   (batched over B*H: M=N=1024, K=64)
    gemm_bf16_kernel<3, false><<<dim3(Tc / 64, Tc / 256, Bc * Hc), 256, 0, stream>>>(
        q_bf, (long long)Tc * Dc, Dc, k_bf, (long long)Tc * Dc, Dc,
        S, (long long)Tc * Tc, Tc, nullptr, nullptr, fire, Tc, nullptr,
        Tc, Tc, Dc, scale);

    softmax_kernel<<<Bc * Hc * Tc, 256, 0, stream>>>(S, P);

    // y = P @ V   (batched: M=1024, N=64, K=1024; Bt = V^T stored D x T)
    gemm_bf16_kernel<0, false><<<dim3(1, Tc / 256, Bc * Hc), 256, 0, stream>>>(
        P, (long long)Tc * Tc, Tc, vT_bf, (long long)Tc * Dc, Tc,
        y, (long long)Tc * Dc, Dc, nullptr, nullptr, nullptr, 0, nullptr,
        Tc, Dc, Tc, 1.0f);

    merge_heads_kernel<<<btc_blocks, 256, 0, stream>>>(y, y_bf);

    // x = x + y @ proj_w + proj_b   (M=2048, N=768, K=768; in place over x)
    gemm_bf16_kernel<1, false><<<dim3(Cc / 64, rows / 256, 1), 256, 0, stream>>>(
        y_bf, 0, Cc, pwT, 0, Cc, x, 0, Cc, nullptr,
        proj_b + (size_t)l * Cc, nullptr, 0, x, rows, Cc, Cc, 1.0f);

    // h2 = LN2(x) -> bf16
    ln_kernel<<<rows, 256, 0, stream>>>(x, ln2_g + (size_t)l * Cc, ln2_b + (size_t)l * Cc, h_bf);

    // u = gelu(h2 @ fc_w + fc_b) -> bf16   (M=2048, N=3072, K=768)
    gemm_bf16_kernel<2, false><<<dim3(4 * Cc / 64, rows / 256, 1), 256, 0, stream>>>(
        h_bf, 0, Cc, fwT, 0, Cc, nullptr, 0, 4 * Cc, u_bf,
        fc_b + (size_t)l * 4 * Cc, nullptr, 0, nullptr, rows, 4 * Cc, Cc, 1.0f);

    // x = x + u @ fc2_w + fc2_b   (M=2048, N=768, K=3072; in place)
    gemm_bf16_kernel<1, false><<<dim3(Cc / 64, rows / 256, 1), 256, 0, stream>>>(
        u_bf, 0, 4 * Cc, f2T, 0, 4 * Cc, x, 0, Cc, nullptr,
        fc2_b + (size_t)l * Cc, nullptr, 0, x, rows, Cc, 4 * Cc, 1.0f);
  }

  // final LN + tied lm head: logits = LNf(x) @ wte^T   (N=50257 ragged)
  ln_kernel<<<rows, 256, 0, stream>>>(x, lnf_g, lnf_b, h_bf);
  gemm_bf16_kernel<0, true><<<dim3((Vc + 63) / 64, rows / 256, 1), 256, 0, stream>>>(
      h_bf, 0, Cc, wte_bf, 0, Cc, out, 0, Vc, nullptr,
      nullptr, nullptr, 0, nullptr, rows, Vc, Cc, 1.0f);
}
